// RRNetModel_41291815584252
// MI455X (gfx1250) — matrix operand order repaired and checked by reading, not compiled, not run on hardware
//
#include <hip/hip_runtime.h>
#include <hip/hip_bf16.h>
#include <math.h>

// ---------------------------------------------------------------------------
// Model dims
// ---------------------------------------------------------------------------
#define kT    128
#define kB    32
#define kH    512            // NHID == NINP
#define kG    2048           // 4*NHID (gate width)
#define kTOK  32000
#define kAK   1536           // activation row: [x(512) | h(512) | stack_top(512)]

// ---------------------------------------------------------------------------
// WMMA fragment types (gfx1250, wave32)
// ---------------------------------------------------------------------------
typedef __attribute__((ext_vector_type(16))) __bf16 v16bf;
typedef __attribute__((ext_vector_type(8)))  float  v8f;

union FragAB { v16bf v; uint4 q[2]; };
union FragC  { v8f   v; float f[8]; };

__device__ __forceinline__ unsigned short f2bf(float x) {
  unsigned u = __float_as_uint(x);
  unsigned r = (u + 0x7FFFu + ((u >> 16) & 1u)) >> 16;   // round-to-nearest-even
  return (unsigned short)r;
}
__device__ __forceinline__ float sigm(float x) { return 1.0f / (1.0f + __expf(-x)); }
__device__ __forceinline__ unsigned mixh(unsigned h) {
  h ^= h >> 16; h *= 0x7FEB352Du; h ^= h >> 15; h *= 0x846CA68Bu; h ^= h >> 16; return h;
}

// ---------------------------------------------------------------------------
// gfx1250 async global->LDS copy (ASYNCcnt path), with graceful fallback
// Builtin signature (probe-confirmed via round-2 diagnostic):
//   void __builtin_amdgcn_global_load_async_to_lds_b128(
//       v4i addrspace(1)* src, v4i addrspace(3)* dst, imm int off, imm int cpol)
// ---------------------------------------------------------------------------
typedef int v4i_t __attribute__((vector_size(16)));
typedef __attribute__((address_space(1))) v4i_t* gv4i_p;
typedef __attribute__((address_space(3))) v4i_t* lv4i_p;

#if defined(__HIP_DEVICE_COMPILE__) && defined(__has_builtin)
#if __has_builtin(__builtin_amdgcn_global_load_async_to_lds_b128)
#define HAVE_ASYNC_COPY 1
#endif
#if __has_builtin(__builtin_amdgcn_s_wait_asynccnt)
#define HAVE_ASYNC_WAIT 1
#endif
#endif
#ifndef HAVE_ASYNC_COPY
#define HAVE_ASYNC_COPY 0
#endif
#ifndef HAVE_ASYNC_WAIT
#define HAVE_ASYNC_WAIT 0
#endif

// src: generic pointer to global memory; lds_generic: generic pointer into a
// __shared__ array (low 32 bits of a generic LDS address == LDS byte offset).
__device__ __forceinline__ void copy16_g2lds(const unsigned short* src,
                                             unsigned short* lds_generic) {
#if HAVE_ASYNC_COPY
  __builtin_amdgcn_global_load_async_to_lds_b128(
      (gv4i_p)(unsigned long long)src,
      (lv4i_p)(unsigned)(unsigned long long)lds_generic,
      0, 0);
#else
  *(uint4*)lds_generic = *(const uint4*)src;   // sync fallback: ds_store_b128
#endif
}

__device__ __forceinline__ void wait_async0() {
#if defined(__HIP_DEVICE_COMPILE__)
#if HAVE_ASYNC_WAIT
  __builtin_amdgcn_s_wait_asynccnt(0);
#elif HAVE_ASYNC_COPY
  asm volatile("s_wait_asynccnt 0x0" ::: "memory");
#endif
#endif
}

// ---------------------------------------------------------------------------
// Generic direct-load GEMM: C[M,N] = A[M,K](bf16) * B[N,K](bf16)^T (+bias).
// One wave = 16x64 strip (4 accumulators). Used for the tiny recurrent-step
// GEMMs (M=32, weights L2-resident, latency-bound).
// ---------------------------------------------------------------------------
__global__ __launch_bounds__(256)
void wmma_gemm_nt(const unsigned short* __restrict__ A, int lda,
                  const unsigned short* __restrict__ B, int ldb,
                  float* __restrict__ C, long long ldc,
                  const float* __restrict__ bias,
                  int M, int N, int K) {
  const int lane  = threadIdx.x & 31;
  const int r     = lane & 15;
  const int hi    = lane >> 4;
  const int koffA = hi * 8;    // A frag: lanes16-31 hold K = +8.. / +24..
  const int koffB = hi * 8;    // B frag: lanes16-31 hold K = +8.. / +24..
  const int nwaves = (gridDim.x * blockDim.x) >> 5;
  const int mt_n = M >> 4;
  const int nq_n = N >> 6;
  const int units = mt_n * nq_n;

  for (int u = (blockIdx.x * blockDim.x + threadIdx.x) >> 5; u < units; u += nwaves) {
    const int mt = u % mt_n;
    const int nq = u / mt_n;
    const unsigned short* Arow = A + (size_t)(mt * 16 + r) * lda;

    FragC acc[4];
#pragma unroll
    for (int s = 0; s < 4; ++s)
#pragma unroll
      for (int e = 0; e < 8; ++e) acc[s].f[e] = 0.0f;

    for (int k0 = 0; k0 < K; k0 += 32) {
      FragAB a;
      a.q[0] = *(const uint4*)(Arow + k0 + koffA);
      a.q[1] = *(const uint4*)(Arow + k0 + koffA + 16);
      if (k0 + 32 < K) __builtin_prefetch((const void*)(Arow + k0 + 32), 0, 1);
#pragma unroll
      for (int s = 0; s < 4; ++s) {
        const unsigned short* Brow =
            B + (size_t)(nq * 64 + s * 16 + r) * ldb + k0 + koffB;
        FragAB b;
        b.q[0] = *(const uint4*)(Brow);
        b.q[1] = *(const uint4*)(Brow + 16);
        acc[s].v = __builtin_amdgcn_wmma_f32_16x16x32_bf16(
            false, a.v, false, b.v, (short)0, acc[s].v, false, false);
      }
    }
#pragma unroll
    for (int s = 0; s < 4; ++s) {
      const int col = nq * 64 + s * 16 + r;
      const float bv = bias ? bias[col] : 0.0f;
#pragma unroll
      for (int v = 0; v < 8; ++v) {
        const int row = mt * 16 + v + hi * 8;     // C/D: M = vgpr + 8*(lane>=16)
        C[(size_t)row * (size_t)ldc + col] = acc[s].f[v] + bv;
      }
    }
  }
}

// ---------------------------------------------------------------------------
// LDS-staged GEMM for the big tail GEMMs (head + decoder).
// Block = 256 threads = 8 waves, computes a 256x64 tile of C.
// B k-chunk (64 rows x 32 k) is staged in LDS via async global->LDS b128
// copies (double-buffered; copy of chunk k+1 overlaps WMMAs of chunk k).
// Each wave: 2 M-tiles x 4 N-tiles = 8 accumulators; B frag read twice from
// LDS (ds_load_b128, rows padded to 80B to spread banks), A frag 4x reuse.
// Requires: M % 256 == 0, N % 64 == 0, K % 32 == 0.
// ---------------------------------------------------------------------------
#define BPAD 40   // LDS row pitch in shorts (80 B)

__global__ __launch_bounds__(256)
void wmma_gemm_lds(const unsigned short* __restrict__ A, int lda,
                   const unsigned short* __restrict__ B, int ldb,
                   float* __restrict__ C, long long ldc,
                   const float* __restrict__ bias,
                   int M, int N, int K) {
  __shared__ unsigned short tileB[2][64][BPAD];

  const int tid  = threadIdx.x;
  const int lane = tid & 31;
  const int wv   = tid >> 5;          // wave 0..7
  const int r    = lane & 15;
  const int hi   = lane >> 4;
  const int koffA = hi * 8;
  const int koffB = hi * 8;

  const int mb_n = M >> 8;
  const int um = blockIdx.x % mb_n;
  const int nq = blockIdx.x / mb_n;
  const int rowBase = um * 256 + wv * 32;

  // cooperative B-stage addressing: 256 threads cover 64 rows x 4 x 16B
  const int ldRow = tid >> 2;          // 0..63
  const int ldSeg = tid & 3;           // 0..3
  const unsigned short* Bsrc0 =
      B + (size_t)(nq * 64 + ldRow) * ldb + ldSeg * 8;

  FragC acc[2][4];
#pragma unroll
  for (int m = 0; m < 2; ++m)
#pragma unroll
    for (int s = 0; s < 4; ++s)
#pragma unroll
      for (int e = 0; e < 8; ++e) acc[m][s].f[e] = 0.0f;

  const int KC = K >> 5;

  // prologue: chunk 0 -> buf 0
  copy16_g2lds(Bsrc0, &tileB[0][ldRow][ldSeg * 8]);

  for (int kc = 0; kc < KC; ++kc) {
    const int buf = kc & 1;
    const int k0 = kc << 5;

    wait_async0();          // this wave's staged chunk is in LDS
    __syncthreads();        // ... and everyone else's

    if (kc + 1 < KC)        // overlap next chunk's copy with this chunk's math
      copy16_g2lds(Bsrc0 + (k0 + 32), &tileB[buf ^ 1][ldRow][ldSeg * 8]);

    FragAB a0, a1;
    const unsigned short* Ar0 = A + (size_t)(rowBase + r) * lda + k0 + koffA;
    const unsigned short* Ar1 = Ar0 + (size_t)16 * lda;
    a0.q[0] = *(const uint4*)(Ar0);
    a0.q[1] = *(const uint4*)(Ar0 + 16);
    a1.q[0] = *(const uint4*)(Ar1);
    a1.q[1] = *(const uint4*)(Ar1 + 16);

#pragma unroll
    for (int s = 0; s < 4; ++s) {
      FragAB b;
      const unsigned short* Bl = &tileB[buf][s * 16 + r][koffB];
      b.q[0] = *(const uint4*)(Bl);        // ds_load_b128
      b.q[1] = *(const uint4*)(Bl + 16);
      acc[0][s].v = __builtin_amdgcn_wmma_f32_16x16x32_bf16(
          false, a0.v, false, b.v, (short)0, acc[0][s].v, false, false);
      acc[1][s].v = __builtin_amdgcn_wmma_f32_16x16x32_bf16(
          false, a1.v, false, b.v, (short)0, acc[1][s].v, false, false);
    }
    __syncthreads();        // readers done before buf is overwritten next round
  }

#pragma unroll
  for (int m = 0; m < 2; ++m)
#pragma unroll
    for (int s = 0; s < 4; ++s) {
      const int col = nq * 64 + s * 16 + r;
      const float bv = bias ? bias[col] : 0.0f;
#pragma unroll
      for (int v = 0; v < 8; ++v) {
        const int row = rowBase + m * 16 + v + hi * 8;
        C[(size_t)row * (size_t)ldc + col] = acc[m][s].f[v] + bv;
      }
    }
}

// ---------------------------------------------------------------------------
// Weight repack kernels (fp32 -> bf16, with K-concatenation)
// ---------------------------------------------------------------------------
__global__ void cvt_cat2(unsigned short* dst, const float* Wa, const float* Wb) {
  const int total = kG * 1024;
  for (int i = blockIdx.x * blockDim.x + threadIdx.x; i < total;
       i += gridDim.x * blockDim.x) {
    int n = i >> 10, c = i & 1023;
    float v = (c < 512) ? Wa[n * 512 + c] : Wb[n * 512 + (c - 512)];
    dst[i] = f2bf(v);
  }
}
__global__ void cvt_m(unsigned short* dst, const float* Wih, const float* Whh) {
  const int total = kG * kAK;
  for (int i = blockIdx.x * blockDim.x + threadIdx.x; i < total;
       i += gridDim.x * blockDim.x) {
    int n = i / kAK, c = i % kAK;
    float v;
    if (c < 512)       v = Wih[n * 1024 + c];
    else if (c < 1024) v = Whh[n * 512 + (c - 512)];
    else               v = Wih[n * 1024 + (c - 1024) + 512];
    dst[i] = f2bf(v);
  }
}
__global__ void cvt_pi(unsigned short* dst, const float* W) {
  const int total = kH * 1024;
  for (int i = blockIdx.x * blockDim.x + threadIdx.x; i < total;
       i += gridDim.x * blockDim.x) {
    int n = i >> 10, c = i & 1023;
    float v = (c < 512) ? W[n * 1024 + 512 + c] : W[n * 1024 + (c - 512)];
    dst[i] = f2bf(v);
  }
}
__global__ void cvt_plain(unsigned short* dst, const float* src, long long total) {
  for (long long i = blockIdx.x * (long long)blockDim.x + threadIdx.x; i < total;
       i += (long long)gridDim.x * blockDim.x)
    dst[i] = f2bf(src[i]);
}

// ---------------------------------------------------------------------------
// Setup: h=h0, c=c0, stack=0, slen=0, first activation tile A0=[emb[0]|h0|0]
// ---------------------------------------------------------------------------
__global__ void setup_kernel(unsigned short* Abf, float* h, float* c,
                             float* sbuf, int* slen,
                             const float* embW, const int* input,
                             const float* h0, const float* c0) {
  int idx = blockIdx.x * blockDim.x + threadIdx.x;
  int stride = gridDim.x * blockDim.x;
  for (int i = idx; i < kB * kAK; i += stride) {
    int b = i / kAK, cix = i % kAK;
    float v = 0.0f;
    if (cix < 512)       { int tok = input[b]; v = embW[(size_t)tok * kH + cix]; }
    else if (cix < 1024) { v = h0[b * kH + (cix - 512)]; }
    Abf[i] = f2bf(v);
  }
  for (int i = idx; i < kB * kH; i += stride) {
    h[i] = h0[i]; c[i] = c0[i];
    sbuf[i] = 0.0f; sbuf[kB * kH + i] = 0.0f;
  }
  if (idx < kB) slen[idx] = 0;
}

// ---------------------------------------------------------------------------
// Per-step fused pointwise: BN(policy) -> logits -> masked Gumbel argmax ->
// 3 LSTM cells -> select -> stack push/pop -> emit next A tile + seq_bf16
// ---------------------------------------------------------------------------
__device__ __forceinline__ void lstm_elem(const float* __restrict__ g,
                                          const float* __restrict__ bi,
                                          const float* __restrict__ bh,
                                          int b, int n, float cc,
                                          float& ho, float& co) {
  float i_ = g[b * kG + n]        + bi[n]        + bh[n];
  float f_ = g[b * kG + 512 + n]  + bi[512 + n]  + bh[512 + n];
  float gg = g[b * kG + 1024 + n] + bi[1024 + n] + bh[1024 + n];
  float o_ = g[b * kG + 1536 + n] + bi[1536 + n] + bh[1536 + n];
  co = sigm(f_) * cc + sigm(i_) * tanhf(gg);
  ho = sigm(o_) * tanhf(co);
}

__global__ __launch_bounds__(1024)
void step_pointwise(int t,
                    const float* __restrict__ gr, const float* __restrict__ gs,
                    const float* __restrict__ gm, const float* __restrict__ zpre,
                    const float* bihR, const float* bhhR,
                    const float* bihS, const float* bhhS,
                    const float* bihM, const float* bhhM,
                    const float* pig, const float* pibe,
                    const float* piW2, const float* pib2,
                    const int* input, const float* embW,
                    float* h, float* c, float* sbuf, int* slen,
                    unsigned short* Abf, unsigned short* seqbf) {
  __shared__ float scale[kH], shift[kH];
  __shared__ int act[kB];
  __shared__ int sls[kB];
  const int tid = threadIdx.x;

  // --- phase 1: BN stats over batch (pi_b1 cancels under BN) ---
  if (tid < kH) {
    float s = 0.f, s2 = 0.f;
    for (int b = 0; b < kB; ++b) {
      float x = zpre[b * kH + tid]; s += x; s2 += x * x;
    }
    float mu = s * (1.f / kB);
    float var = s2 * (1.f / kB) - mu * mu;
    float rs = rsqrtf(var + 1e-5f) * pig[tid];
    scale[tid] = rs; shift[tid] = pibe[tid] - mu * rs;
  }
  if (tid < kB) sls[tid] = slen[tid];
  __syncthreads();

  // --- phase 2: logits + masked Gumbel-argmax action ---
  if (tid < kB) {
    const int b = tid;
    float lg[3] = {pib2[0], pib2[1], pib2[2]};
    for (int k = 0; k < kH; ++k) {
      float z = zpre[b * kH + k] * scale[k] + shift[k];
      z = z > 0.f ? z : 0.f;
      lg[0] += z * piW2[k];
      lg[1] += z * piW2[512 + k];
      lg[2] += z * piW2[1024 + k];
    }
    const int sl = sls[b];
    bool dead0 = (sl >= 2);   // push masked when stack full
    bool dead2 = (sl == 0);   // pop masked when stack empty
    float best = -3.0e38f; int ai = 1;
    for (int j = 0; j < 3; ++j) {
      if ((j == 0 && dead0) || (j == 2 && dead2)) continue;
      unsigned hh = mixh(0x2545F491u ^ (unsigned)(t * 1315423911)
                         ^ (unsigned)(b * 2654435761u) ^ (unsigned)(j + 42));
      float u = ((hh >> 8) + 0.5f) * (1.f / 16777216.f);
      float gn = -logf(-logf(u));
      float v = lg[j] + gn;
      if (v > best) { best = v; ai = j; }
    }
    act[b] = ai;
  }
  __syncthreads();

  // --- phase 3: LSTM cells, select, stack, next A tile, seq output ---
  for (int e = tid; e < kB * kH; e += 1024) {
    const int b = e >> 9, n = e & 511;
    const int a = act[b];
    const int sl = sls[b];
    const float cc = c[e];
    float hr, cr, hs, cs, hm, cm;
    lstm_elem(gr, bihR, bhhR, b, n, cc, hr, cr);
    lstm_elem(gs, bihS, bhhS, b, n, cc, hs, cs);
    lstm_elem(gm, bihM, bhhM, b, n, cc, hm, cm);
    const float nh = (a == 0) ? hs : ((a == 1) ? hr : hm);
    const float nc = (a == 0) ? cs : ((a == 1) ? cr : cm);
    h[e] = nh; c[e] = nc;
    seqbf[(size_t)(t * kB + b) * kH + n] = f2bf(nh);
    if (a == 0) {
      const int pidx = sl > 1 ? 1 : sl;           // clip(slen, 0, SMAX-1)
      sbuf[(b * 2 + pidx) * kH + n] = hs;
    }
    const int nsl = sl + ((a == 0) ? 1 : ((a == 2) ? -1 : 0));
    float si = 0.0f;
    if (a == 0) si = hs;                          // just-pushed top
    else if (nsl > 0) si = sbuf[(b * 2 + (nsl - 1)) * kH + n];
    if (t + 1 < kT) {
      const int tok = input[(t + 1) * kB + b];
      Abf[b * kAK + n] = f2bf(embW[(size_t)tok * kH + n]);
    } else {
      Abf[b * kAK + n] = 0;
    }
    Abf[b * kAK + 512 + n]  = f2bf(nh);
    Abf[b * kAK + 1024 + n] = f2bf(si);
  }
  __syncthreads();

  // --- phase 4: stack-length update ---
  if (tid < kB) {
    const int a = act[tid];
    slen[tid] = sls[tid] + ((a == 0) ? 1 : ((a == 2) ? -1 : 0));
  }
}

// ---------------------------------------------------------------------------
// Head BN + tanh over the flattened sequence (per-column, 4096-row stats)
// ---------------------------------------------------------------------------
__global__ __launch_bounds__(256)
void bn_tanh_kernel(const float* __restrict__ X, const float* g,
                    const float* be, unsigned short* __restrict__ out) {
  __shared__ float red[256];
  const int j = blockIdx.x;           // column 0..511
  const int tid = threadIdx.x;
  const int rows = kT * kB;           // 4096
  float s = 0.f, s2 = 0.f;
  for (int i = tid; i < rows; i += 256) {
    float x = X[(size_t)i * kH + j]; s += x; s2 += x * x;
  }
  red[tid] = s; __syncthreads();
  for (int o = 128; o > 0; o >>= 1) {
    if (tid < o) red[tid] += red[tid + o];
    __syncthreads();
  }
  const float mu = red[0] * (1.f / rows);
  __syncthreads();
  red[tid] = s2; __syncthreads();
  for (int o = 128; o > 0; o >>= 1) {
    if (tid < o) red[tid] += red[tid + o];
    __syncthreads();
  }
  const float var = red[0] * (1.f / rows) - mu * mu;
  const float rs = rsqrtf(var + 1e-5f) * g[j];
  const float sh = be[j] - mu * rs;
  for (int i = tid; i < rows; i += 256) {
    float x = X[(size_t)i * kH + j];
    out[(size_t)i * kH + j] = f2bf(tanhf(x * rs + sh));
  }
}

__global__ void tail_copy(const float* h, const float* c, float* out) {
  int i = blockIdx.x * blockDim.x + threadIdx.x;
  if (i < kB * kH) { out[i] = h[i]; out[kB * kH + i] = c[i]; }
}

// ---------------------------------------------------------------------------
// Launcher
// ---------------------------------------------------------------------------
extern "C" void kernel_launch(void* const* d_in, const int* in_sizes, int n_in,
                              void* d_out, int out_size, void* d_ws, size_t ws_size,
                              hipStream_t stream) {
  const int*   input = (const int*)  d_in[0];
  const float* h0    = (const float*)d_in[1];
  const float* c0    = (const float*)d_in[2];
  const float* embW  = (const float*)d_in[3];
  const float* decW  = (const float*)d_in[4];
  const float* decb  = (const float*)d_in[5];
  const float* WihR  = (const float*)d_in[6];
  const float* WhhR  = (const float*)d_in[7];
  const float* bihR  = (const float*)d_in[8];
  const float* bhhR  = (const float*)d_in[9];
  const float* WihS  = (const float*)d_in[10];
  const float* WhhS  = (const float*)d_in[11];
  const float* bihS  = (const float*)d_in[12];
  const float* bhhS  = (const float*)d_in[13];
  const float* WihM  = (const float*)d_in[14];
  const float* WhhM  = (const float*)d_in[15];
  const float* bihM  = (const float*)d_in[16];
  const float* bhhM  = (const float*)d_in[17];
  const float* piW1  = (const float*)d_in[18];
  // d_in[19] = pi_b1 : cancels inside BN, unused
  const float* pig   = (const float*)d_in[20];
  const float* pibe  = (const float*)d_in[21];
  const float* piW2  = (const float*)d_in[22];
  const float* pib2  = (const float*)d_in[23];
  const float* prW   = (const float*)d_in[24];
  // d_in[25] = pr_b : cancels inside BN, unused
  const float* prg   = (const float*)d_in[26];
  const float* prbe  = (const float*)d_in[27];

  // ---- workspace carve-out (256B aligned) ----
  char* ws = (char*)d_ws;
  size_t off = 0;
  auto take = [&](size_t bytes) -> char* {
    char* p = ws + off; off += (bytes + 255) & ~(size_t)255; return p;
  };
  unsigned short* Wr   = (unsigned short*)take((size_t)kG * 1024 * 2);
  unsigned short* Ws   = (unsigned short*)take((size_t)kG * 1024 * 2);
  unsigned short* Wm   = (unsigned short*)take((size_t)kG * kAK * 2);
  unsigned short* Wpi  = (unsigned short*)take((size_t)kH * 1024 * 2);
  unsigned short* Wpr  = (unsigned short*)take((size_t)kH * kH * 2);
  unsigned short* Wdec = (unsigned short*)take((size_t)kTOK * kH * 2);
  unsigned short* Abf  = (unsigned short*)take((size_t)kB * kAK * 2);
  float* h    = (float*)take((size_t)kB * kH * 4);
  float* c    = (float*)take((size_t)kB * kH * 4);
  float* sbuf = (float*)take((size_t)kB * 2 * kH * 4);
  int*   slen = (int*)  take((size_t)kB * 4);
  float* gr   = (float*)take((size_t)kB * kG * 4);
  float* gs   = (float*)take((size_t)kB * kG * 4);
  float* gm   = (float*)take((size_t)kB * kG * 4);
  float* zpre = (float*)take((size_t)kB * kH * 4);
  unsigned short* seqbf  = (unsigned short*)take((size_t)kT * kB * kH * 2);
  float*          predpre= (float*)take((size_t)kT * kB * kH * 4);
  unsigned short* predbf = (unsigned short*)take((size_t)kT * kB * kH * 2);
  (void)ws_size; (void)in_sizes; (void)n_in; (void)out_size;

  // ---- weight repack (bf16, K-concatenated) ----
  cvt_cat2 <<<512, 256, 0, stream>>>(Wr, WihR, WhhR);
  cvt_cat2 <<<512, 256, 0, stream>>>(Ws, WihS, WhhS);
  cvt_m    <<<512, 256, 0, stream>>>(Wm, WihM, WhhM);
  cvt_pi   <<<512, 256, 0, stream>>>(Wpi, piW1);
  cvt_plain<<<512, 256, 0, stream>>>(Wpr, prW, (long long)kH * kH);
  cvt_plain<<<2048, 256, 0, stream>>>(Wdec, decW, (long long)kTOK * kH);

  setup_kernel<<<192, 256, 0, stream>>>(Abf, h, c, sbuf, slen, embW, input, h0, c0);

  auto gemm = [&](const unsigned short* A, int lda, const unsigned short* Bm,
                  int ldb, float* C, long long ldc, const float* bias,
                  int M, int N, int K) {
    int units = (M >> 4) * (N >> 6);
    int blocks = (units + 7) / 8;             // 8 waves per 256-thread block
    wmma_gemm_nt<<<blocks, 256, 0, stream>>>(A, lda, Bm, ldb, C, ldc, bias, M, N, K);
  };
  auto gemm_big = [&](const unsigned short* A, int lda, const unsigned short* Bm,
                      int ldb, float* C, long long ldc, const float* bias,
                      int M, int N, int K) {
    int blocks = (M >> 8) * (N >> 6);         // 256x64 C-tile per block
    wmma_gemm_lds<<<blocks, 256, 0, stream>>>(A, lda, Bm, ldb, C, ldc, bias, M, N, K);
  };

  // ---- recurrent scan: 4 WMMA GEMMs + fused pointwise per step ----
  for (int t = 0; t < kT; ++t) {
    gemm(Abf, kAK, Wr,  1024, gr,   kG, nullptr, kB, kG, 1024);
    gemm(Abf, kAK, Ws,  1024, gs,   kG, nullptr, kB, kG, 1024);
    gemm(Abf, kAK, Wm,  kAK,  gm,   kG, nullptr, kB, kG, kAK);
    gemm(Abf, kAK, Wpi, 1024, zpre, kH, nullptr, kB, kH, 1024);
    step_pointwise<<<1, 1024, 0, stream>>>(t, gr, gs, gm, zpre,
                                           bihR, bhhR, bihS, bhhS, bihM, bhhM,
                                           pig, pibe, piW2, pib2,
                                           input, embW,
                                           h, c, sbuf, slen, Abf, seqbf);
  }

  // ---- head: pred = tanh(BN(seq @ pr_W^T)) ----
  gemm_big(seqbf, kH, Wpr, kH, predpre, kH, nullptr, kT * kB, kH, kH);
  bn_tanh_kernel<<<kH, 256, 0, stream>>>(predpre, prg, prbe, predbf);

  // ---- decoder: d_out[0 : 4096*32000] = pred @ dec_W^T + dec_b ----
  float* dec = (float*)d_out;
  gemm_big(predbf, kH, Wdec, kH, dec, kTOK, decb, kT * kB, kTOK, kH);

  // ---- hT, cT appended after dec ----
  tail_copy<<<64, 256, 0, stream>>>(h, c, dec + (size_t)kT * kB * kTOK);
}
